// SparseMoE_78606491451884
// MI455X (gfx1250) — matrix operand order repaired and checked by reading, not compiled, not run on hardware
//
#include <hip/hip_runtime.h>
#include <math.h>

// ---------------- problem constants ----------------
#define NTOK   16384
#define DMODEL 1024
#define DFF    4096
#define NEXP   8
#define TOPK   2

// ---------------- tiling ----------------
#define ROWS    32      // routed rows per workgroup
#define CHUNK   64      // DFF chunk width
#define KBLK    128     // D sub-block staged per phase-A pipeline step
#define THREADS 256     // 8 wave32 per workgroup

// LDS leading dims (halves); pads keep 16B alignment & break bank conflicts
#define XS_LD  (DMODEL + 8)   // 1032
#define W1T_LD (KBLK + 8)     // 136  (w1t[dff_col][k over D]
#define W2T_LD (CHUNK + 8)    // 72   (w2t[d_col][k over DFF]
#define HS_LD  (CHUNK + 8)    // 72

#define XS_BYTES  (ROWS * XS_LD * 2)       // 66048
#define W1T_BYTES (CHUNK * W1T_LD * 2)     // 17408 per buffer
#define W2T_BYTES (KBLK * W2T_LD * 2)      // 18432 per buffer
#define HS_BYTES  (ROWS * HS_LD * 2)       // 4608

#define OFF_XS   0
#define OFF_W1T0 (OFF_XS + XS_BYTES)
#define OFF_W1T1 (OFF_W1T0 + W1T_BYTES)
#define OFF_W2T0 (OFF_W1T1 + W1T_BYTES)
#define OFF_W2T1 (OFF_W2T0 + W2T_BYTES)
#define OFF_HS   (OFF_W2T1 + W2T_BYTES)
#define OFF_TOKI (OFF_HS + HS_BYTES)
#define OFF_TOKW (OFF_TOKI + 128)
#define SMEM_TOTAL (OFF_TOKW + 128)        // 142592

typedef __bf16 bf16_t;
typedef __attribute__((ext_vector_type(4)))  __bf16 v4bf;
typedef __attribute__((ext_vector_type(8)))  __bf16 v8bf;
typedef __attribute__((ext_vector_type(16))) __bf16 v16bf;
typedef __attribute__((ext_vector_type(8)))  float  v8f;

// ---------------- WMMA helper ----------------
__device__ __forceinline__ v8f wmma_bf16(v16bf a, v16bf b, v8f c) {
  return __builtin_amdgcn_wmma_f32_16x16x32_bf16(
      false, a, false, b, (short)0, c, false, false);
}

// A-matrix 16x32 bf16 fragment from row-major LDS tile.
// lanes 0-15: row M=l, K 0..7 (v0..3) and 16..23 (v4..7); lanes 16-31: K 8..15, 24..31.
__device__ __forceinline__ v16bf load_frag_A(const bf16_t* tile, int ld,
                                             int rowBase, int kBase,
                                             int l15, int hi) {
  const bf16_t* p = tile + (size_t)(rowBase + l15) * ld + kBase + hi * 8;
  v8bf lo = *(const v8bf*)p;
  v8bf up = *(const v8bf*)(p + 16);
  v16bf r;
#pragma unroll
  for (int i = 0; i < 8; ++i) { r[i] = lo[i]; r[i + 8] = up[i]; }
  return r;
}

// B-matrix 32x16 bf16 fragment from transposed LDS tile WT[col][K].
// lanes 0-15: column N=l, K 0..7 and 16..23; lanes 16-31: K 8..15, 24..31.
__device__ __forceinline__ v16bf load_frag_B(const bf16_t* tileT, int ld,
                                             int colBase, int kBase,
                                             int l15, int hi) {
  const bf16_t* p = tileT + (size_t)(colBase + l15) * ld + kBase + hi * 8;
  v8bf lo = *(const v8bf*)p;
  v8bf up = *(const v8bf*)(p + 16);
  v16bf r;
#pragma unroll
  for (int i = 0; i < 8; ++i) { r[i] = lo[i]; r[i + 8] = up[i]; }
  return r;
}

__device__ __forceinline__ float gelu_exact(float v) {
  return 0.5f * v * (1.0f + erff(v * 0.70710678118654752f));
}

// ---------------- pipelined weight staging helpers ----------------
// W1 block (kc, kb): KBLK=128 D-rows x CHUNK=64 DFF-cols.
// Thread owns column cA (0..63) and row-quad rqA (0..3); coalesced b32 loads
// down the column so the transposed LDS store is one contiguous v4bf.
__device__ __forceinline__ void load_w1_regs(const float* __restrict__ W1,
                                             size_t w1base, int kc, int kb,
                                             int cA, int rqA, float rbuf[32]) {
  const float* base = W1 + w1base + (size_t)kc * CHUNK + cA;
#pragma unroll
  for (int it = 0; it < 8; ++it) {
#pragma unroll
    for (int j = 0; j < 4; ++j) {
      const int r = it * 16 + rqA * 4 + j;
      rbuf[it * 4 + j] = base[(size_t)(kb * KBLK + r) * DFF];
    }
  }
}
__device__ __forceinline__ void store_w1_lds(bf16_t* buf, int cA, int rqA,
                                             const float rbuf[32]) {
  bf16_t* dst = buf + (size_t)cA * W1T_LD + rqA * 4;
#pragma unroll
  for (int it = 0; it < 8; ++it) {
    v4bf p;
#pragma unroll
    for (int j = 0; j < 4; ++j) p[j] = (bf16_t)rbuf[it * 4 + j];
    *(v4bf*)(dst + it * 16) = p;
  }
}

// W2 block (kc, cb): CHUNK=64 DFF-rows x KBLK=128 D-cols.
// Thread owns column cB (0..127) and row-quad rqB (0..1).
__device__ __forceinline__ void load_w2_regs(const float* __restrict__ W2,
                                             size_t w2base, int kc, int cb,
                                             int cB, int rqB, float rbuf[32]) {
  const float* base = W2 + w2base + (size_t)cb * KBLK + cB;
#pragma unroll
  for (int it = 0; it < 8; ++it) {
#pragma unroll
    for (int j = 0; j < 4; ++j) {
      const int r = it * 8 + rqB * 4 + j;
      rbuf[it * 4 + j] = base[(size_t)(kc * CHUNK + r) * DMODEL];
    }
  }
}
__device__ __forceinline__ void store_w2_lds(bf16_t* buf, int cB, int rqB,
                                             const float rbuf[32]) {
  bf16_t* dst = buf + (size_t)cB * W2T_LD + rqB * 4;
#pragma unroll
  for (int it = 0; it < 8; ++it) {
    v4bf p;
#pragma unroll
    for (int j = 0; j < 4; ++j) p[j] = (bf16_t)rbuf[it * 4 + j];
    *(v4bf*)(dst + it * 8) = p;
  }
}

// ---------------- kernel 0: zero out + counters ----------------
__global__ void moe_zero_kernel(float* __restrict__ out, int* __restrict__ cnt) {
  size_t i = (size_t)blockIdx.x * blockDim.x + threadIdx.x;
  const size_t total = (size_t)NTOK * DMODEL;
  for (size_t j = i; j < total; j += (size_t)gridDim.x * blockDim.x)
    out[j] = 0.0f;
  if (i < (size_t)NEXP) cnt[i] = 0;
}

// ---------------- kernel 1: gating (one wave per token) ----------------
__global__ void moe_gate_kernel(const float* __restrict__ x,
                                const float* __restrict__ gW,
                                const float* __restrict__ gb,
                                int* __restrict__ cnt,
                                int* __restrict__ bTok,
                                float* __restrict__ bW,
                                int* __restrict__ topi) {
  const int gtid = blockIdx.x * blockDim.x + threadIdx.x;
  const int tok  = gtid >> 5;
  const int lane = threadIdx.x & 31;
  if (tok >= NTOK) return;

  const float* xr = x + (size_t)tok * DMODEL;
  float acc[NEXP];
#pragma unroll
  for (int e = 0; e < NEXP; ++e) acc[e] = 0.0f;

  for (int i = 0; i < DMODEL / 32; ++i) {
    const int d = lane + i * 32;
    const float xv = xr[d];
    const float4* g = (const float4*)(gW + (size_t)d * NEXP);
    const float4 g0 = g[0];
    const float4 g1 = g[1];
    acc[0] += xv * g0.x; acc[1] += xv * g0.y;
    acc[2] += xv * g0.z; acc[3] += xv * g0.w;
    acc[4] += xv * g1.x; acc[5] += xv * g1.y;
    acc[6] += xv * g1.z; acc[7] += xv * g1.w;
  }
#pragma unroll
  for (int off = 16; off > 0; off >>= 1) {
#pragma unroll
    for (int e = 0; e < NEXP; ++e) acc[e] += __shfl_xor(acc[e], off, 32);
  }

  if (lane == 0) {
    float logit[NEXP];
#pragma unroll
    for (int e = 0; e < NEXP; ++e) logit[e] = acc[e] + gb[e];
    int i0 = 0; float v0 = logit[0];
#pragma unroll
    for (int e = 1; e < NEXP; ++e) if (logit[e] > v0) { v0 = logit[e]; i0 = e; }
    int i1 = -1; float v1 = -3.4e38f;
#pragma unroll
    for (int e = 0; e < NEXP; ++e)
      if (e != i0 && logit[e] > v1) { v1 = logit[e]; i1 = e; }
    const float e1  = __expf(v1 - v0);
    const float inv = 1.0f / (1.0f + e1);

    int p0 = atomicAdd(&cnt[i0], 1);
    bTok[(size_t)i0 * NTOK + p0] = tok;
    bW  [(size_t)i0 * NTOK + p0] = inv;
    int p1 = atomicAdd(&cnt[i1], 1);
    bTok[(size_t)i1 * NTOK + p1] = tok;
    bW  [(size_t)i1 * NTOK + p1] = e1 * inv;

    topi[tok * TOPK + 0] = i0;
    topi[tok * TOPK + 1] = i1;
  }
}

// ---------------- kernel 2: fused expert FFN (WMMA bf16, pipelined) ----------------
__global__ __launch_bounds__(THREADS, 1)
void moe_ffn_kernel(const float* __restrict__ x,
                    const float* __restrict__ W1,
                    const float* __restrict__ b1,
                    const float* __restrict__ W2,
                    const float* __restrict__ b2,
                    const int* __restrict__ cnt,
                    const int* __restrict__ bTok,
                    const float* __restrict__ bW,
                    float* __restrict__ out) {
  const int e        = blockIdx.y;
  const int rowStart = blockIdx.x * ROWS;
  const int nRouted  = cnt[e];
  if (rowStart >= nRouted) return;

  extern __shared__ char smem[];
  bf16_t* xs   = (bf16_t*)(smem + OFF_XS);
  bf16_t* w1t0 = (bf16_t*)(smem + OFF_W1T0);
  bf16_t* w1t1 = (bf16_t*)(smem + OFF_W1T1);
  bf16_t* w2t0 = (bf16_t*)(smem + OFF_W2T0);
  bf16_t* w2t1 = (bf16_t*)(smem + OFF_W2T1);
  bf16_t* hs   = (bf16_t*)(smem + OFF_HS);
  int*    tokI = (int*)  (smem + OFF_TOKI);
  float*  tokW = (float*)(smem + OFF_TOKW);

  const int tid  = threadIdx.x;
  const int wv   = tid >> 5;
  const int lane = tid & 31;
  const int l15  = lane & 15;
  const int hi   = lane >> 4;
  // phase A ownership: one 16x16 h tile per wave
  const int rtA = wv >> 2;          // h row tile 0..1
  const int ctA = wv & 3;           // h col tile 0..3 (16-wide within 64 chunk)
  // staging lane roles
  const int cA  = tid & 63;         // W1: dff col
  const int rqA = tid >> 6;         // W1: row quad 0..3
  const int cB  = tid & 127;        // W2: d col
  const int rqB = tid >> 7;         // W2: row quad 0..1

  if (tid < ROWS) {
    const int r = rowStart + tid;
    if (r < nRouted) {
      tokI[tid] = bTok[(size_t)e * NTOK + r];
      tokW[tid] = bW  [(size_t)e * NTOK + r];
    } else {
      tokI[tid] = 0;
      tokW[tid] = 0.0f;
    }
  }
  __syncthreads();

  // stage x rows fp32 -> bf16 (gathered by token id), 32 x 1024
  for (int it = 0; it < (ROWS * DMODEL) / (THREADS * 4); ++it) {
    const int linear = (it * THREADS + tid) * 4;
    const int r = linear >> 10;
    const int c = linear & (DMODEL - 1);
    const float4 v = *(const float4*)(x + (size_t)tokI[r] * DMODEL + c);
    v4bf p;
    p[0] = (bf16_t)v.x; p[1] = (bf16_t)v.y;
    p[2] = (bf16_t)v.z; p[3] = (bf16_t)v.w;
    *(v4bf*)(xs + (size_t)r * XS_LD + c) = p;
  }

  const size_t w1base = (size_t)e * DMODEL * DFF;
  const size_t w2base = (size_t)e * DFF * DMODEL;

  v8f acc[16];                       // y tiles: [rt 0..1] x [cb 0..7]
  const v8f vzero = {};
#pragma unroll
  for (int t = 0; t < 16; ++t) acc[t] = vzero;

  float rbuf[32];

  for (int kc = 0; kc < DFF / CHUNK; ++kc) {        // 64 DFF chunks
    // ================= phase A: h(32x64) = x_tile @ W1[:, chunk] =================
    v8f hacc = vzero;
    // pipeline prologue: buf0 <- block0, regs <- block1
    load_w1_regs(W1, w1base, kc, 0, cA, rqA, rbuf);
    store_w1_lds(w1t0, cA, rqA, rbuf);
    load_w1_regs(W1, w1base, kc, 1, cA, rqA, rbuf);
    __syncthreads();                                 // also covers xs staging on kc==0
    for (int kb = 0; kb < DMODEL / KBLK; ++kb) {     // 8 pipeline steps
      bf16_t* cur = (kb & 1) ? w1t1 : w1t0;
      bf16_t* nxt = (kb & 1) ? w1t0 : w1t1;
      if (kb < 7) store_w1_lds(nxt, cA, rqA, rbuf);  // drain regs into other buffer
      if (kb < 6) load_w1_regs(W1, w1base, kc, kb + 2, cA, rqA, rbuf);  // overlap w/ WMMA
#pragma unroll
      for (int ks = 0; ks < KBLK / 32; ++ks) {
        const v16bf bfr = load_frag_B(cur, W1T_LD, ctA * 16, ks * 32, l15, hi);
        const v16bf afr = load_frag_A(xs, XS_LD, rtA * 16, kb * KBLK + ks * 32, l15, hi);
        hacc = wmma_bf16(afr, bfr, hacc);
      }
      __syncthreads();
    }

    // bias + exact GELU -> hs (row-major bf16 32x64)
    {
      const int gdff  = kc * CHUNK + ctA * 16 + l15;
      const float bias = b1[(size_t)e * DFF + gdff];
#pragma unroll
      for (int i = 0; i < 8; ++i) {
        const float hv = gelu_exact(hacc[i] + bias);
        hs[(size_t)(rtA * 16 + hi * 8 + i) * HS_LD + ctA * 16 + l15] = (bf16_t)hv;
      }
    }

    // ================= phase B: y(32x1024) += h_chunk @ W2[chunk, :] =================
    // prologue: buf0 <- block0, regs <- block1 (barrier below also publishes hs)
    load_w2_regs(W2, w2base, kc, 0, cB, rqB, rbuf);
    store_w2_lds(w2t0, cB, rqB, rbuf);
    load_w2_regs(W2, w2base, kc, 1, cB, rqB, rbuf);
    __syncthreads();
#pragma unroll
    for (int cb = 0; cb < DMODEL / KBLK; ++cb) {     // 8 pipeline steps (unrolled: acc idx)
      bf16_t* cur = (cb & 1) ? w2t1 : w2t0;
      bf16_t* nxt = (cb & 1) ? w2t0 : w2t1;
      if (cb < 7) store_w2_lds(nxt, cB, rqB, rbuf);
      if (cb < 6) load_w2_regs(W2, w2base, kc, cb + 2, cB, rqB, rbuf);
#pragma unroll
      for (int ks = 0; ks < CHUNK / 32; ++ks) {
        const v16bf bfr = load_frag_B(cur, W2T_LD, wv * 16, ks * 32, l15, hi);
        const v16bf a0  = load_frag_A(hs, HS_LD, 0,  ks * 32, l15, hi);
        const v16bf a1  = load_frag_A(hs, HS_LD, 16, ks * 32, l15, hi);
        acc[0 * 8 + cb] = wmma_bf16(a0, bfr, acc[0 * 8 + cb]);
        acc[1 * 8 + cb] = wmma_bf16(a1, bfr, acc[1 * 8 + cb]);
      }
      __syncthreads();
    }
  }

  // ----- epilogue: out[token] += w * (y + b2) -----
#pragma unroll
  for (int rt = 0; rt < 2; ++rt) {
#pragma unroll
    for (int cb = 0; cb < 8; ++cb) {
      const int gcol = cb * KBLK + wv * 16 + l15;
      const float bias2 = b2[(size_t)e * DMODEL + gcol];
#pragma unroll
      for (int i = 0; i < 8; ++i) {
        const int row = rt * 16 + hi * 8 + i;        // C/D layout: M = hi*8+i
        const float wt = tokW[row];
        if (wt != 0.0f) {
          const float val = acc[rt * 8 + cb][i] + bias2;
          atomicAdd(out + (size_t)tokI[row] * DMODEL + gcol, wt * val);
        }
      }
    }
  }
}

// ---------------- host entry ----------------
extern "C" void kernel_launch(void* const* d_in, const int* in_sizes, int n_in,
                              void* d_out, int out_size, void* d_ws, size_t ws_size,
                              hipStream_t stream) {
  (void)in_sizes; (void)n_in; (void)out_size; (void)ws_size;

  const float* x  = (const float*)d_in[0];
  const float* gW = (const float*)d_in[1];
  const float* gb = (const float*)d_in[2];
  const float* W1 = (const float*)d_in[3];
  const float* b1 = (const float*)d_in[4];
  const float* W2 = (const float*)d_in[5];
  const float* b2 = (const float*)d_in[6];

  float* outF = (float*)d_out;                                  // [N, D] f32
  int*   topi = (int*)(outF + (size_t)NTOK * DMODEL);           // [N, K] i32

  char*  ws   = (char*)d_ws;
  int*   cnt  = (int*)ws;                                       // [E]
  int*   bTok = (int*)(ws + 128);                               // [E, N]
  float* bW   = (float*)(ws + 128 + (size_t)NEXP * NTOK * 4);   // [E, N]

  moe_zero_kernel<<<4096, 256, 0, stream>>>(outF, cnt);

  moe_gate_kernel<<<(NTOK * 32) / 256, 256, 0, stream>>>(
      x, gW, gb, cnt, bTok, bW, topi);

  dim3 grid(NTOK / ROWS, NEXP);
  moe_ffn_kernel<<<grid, THREADS, SMEM_TOTAL, stream>>>(
      x, W1, b1, W2, b2, cnt, bTok, bW, outF);
}
